// LaneFE_82952998355462
// MI455X (gfx1250) — compile-verified, matrix operand order checked by reading, not run on hardware
//
#include <hip/hip_runtime.h>
#include <hip/hip_bf16.h>
#include <math.h>

typedef __attribute__((ext_vector_type(16))) __bf16 v16bf;
typedef __attribute__((ext_vector_type(8)))  __bf16 v8bf;
typedef __attribute__((ext_vector_type(8)))  float  v8f;

#define SEQ 80
#define BLANES 2560          // BATCH * NUM_MAX_PATHS
#define HDIM 128
#define GDIM 512             // 4*H
#define ROWS_PER_BLOCK 16
#define NTILES 32            // GDIM/16
#define KCHUNKS 4            // HDIM/32
// per-matrix packed fragment storage: NTILES*KCHUNKS*32 lanes*16 bf16 = 65536 elems
#define FRAG_ELEMS_PER_MATRIX 65536

__device__ __forceinline__ float fast_sigmoid(float x) {
    return 1.0f / (1.0f + __expf(-x));
}
__device__ __forceinline__ float fast_tanh(float x) {
    float e = __expf(2.0f * x);
    return (e - 1.0f) / (e + 1.0f);
}

// ---------------------------------------------------------------------------
// Kernel 0: pack W_ih / W_hh (f32 [512,128]) into bf16 WMMA B-fragments.
// Fragment layout (mirrors the 16-bit A 16x32 striping with M->N):
//   lanes 0-15:  N = lane,    K' in {0..7} (slots 0-7) and {16..23} (slots 8-15)
//   lanes 16-31: N = lane-16, K' in {8..15}            and {24..31}
// Storage index = ((tile*KCHUNKS + kc)*32 + lane)*16 + slot
// ---------------------------------------------------------------------------
__global__ void lanefe_pack_weights(const float* __restrict__ Wih,
                                    const float* __restrict__ Whh,
                                    __bf16* __restrict__ frags) {
    int idx = blockIdx.x * blockDim.x + threadIdx.x;       // 0 .. 131071
    const float* W = (idx < FRAG_ELEMS_PER_MATRIX) ? Wih : Whh;
    int e    = idx & (FRAG_ELEMS_PER_MATRIX - 1);
    int tile = e >> 11;            // 2048 elems per N-tile
    int kc   = (e >> 9) & 3;
    int lane = (e >> 4) & 31;
    int slot = e & 15;
    int n  = lane & 15;
    int kh = lane >> 4;
    int kp = (slot < 8) ? (kh * 8 + slot) : (16 + kh * 8 + (slot - 8));
    int g  = tile * 16 + n;        // gate (output) column
    int k  = kc * 32 + kp;         // contraction index
    frags[idx] = (__bf16)W[g * HDIM + k];
}

// ---------------------------------------------------------------------------
// Kernel 1: persistent fused feature + embed + LSTM.
// One block = 16 sequence rows. 8 waves; wave w owns hidden cols [16w,16w+16)
// for all four gates (tiles q*8+w, q=0..3 -> i,f,g,o).
// ---------------------------------------------------------------------------
__global__ void __launch_bounds__(256)
lanefe_lstm_kernel(const float* __restrict__ lanes,   // [SEQ][BLANES][2]
                   const float* __restrict__ Wemb,    // [H][5]
                   const float* __restrict__ bemb,    // [H]
                   const float* __restrict__ bih,     // [4H]
                   const float* __restrict__ bhh,     // [4H]
                   const __bf16* __restrict__ frags,  // packed W_ih | W_hh
                   float* __restrict__ out)           // [BLANES][H]
{
    __shared__ __align__(32) __bf16 emb_s[ROWS_PER_BLOCK][HDIM];
    __shared__ __align__(32) __bf16 h_s[ROWS_PER_BLOCK][HDIM];
    __shared__ float feat_s[ROWS_PER_BLOCK][5];
    __shared__ float wemb_s[HDIM * 5];
    __shared__ float bemb_s[HDIM];
    __shared__ int   valid_s[ROWS_PER_BLOCK];

    const int tid     = threadIdx.x;
    const int lane    = tid & 31;
    const int w       = tid >> 5;                 // wave id, 0..7
    const int rowbase = blockIdx.x * ROWS_PER_BLOCK;

    // Cache embedding weights/bias in LDS (tiny, L2-resident anyway).
    for (int i = tid; i < HDIM * 5; i += 256) wemb_s[i] = Wemb[i];
    for (int i = tid; i < HDIM; i += 256)     bemb_s[i] = bemb[i];

    // h0 = 0
    for (int i = tid; i < ROWS_PER_BLOCK * HDIM; i += 256)
        ((__bf16*)h_s)[i] = (__bf16)0.0f;

    // Per-row NaN validity scan (reference zeroes features of any-NaN lanes).
    if (tid < ROWS_PER_BLOCK) {
        int r = rowbase + tid;
        int ok = 1;
        for (int s = 0; s < SEQ; ++s) {
            float x = lanes[(size_t)(s * BLANES + r) * 2 + 0];
            float y = lanes[(size_t)(s * BLANES + r) * 2 + 1];
            if (x != x || y != y) ok = 0;
        }
        valid_s[tid] = ok;
    }

    // Combined gate bias, pre-loaded into each wave's accumulators.
    float bias[4];
#pragma unroll
    for (int q = 0; q < 4; ++q) {
        int col = q * HDIM + w * 16 + (lane & 15);
        bias[q] = bih[col] + bhh[col];
    }

    float cst[8];                                  // cell state (C-layout rows)
#pragma unroll
    for (int j = 0; j < 8; ++j) cst[j] = 0.0f;

    __syncthreads();

    const int m  = lane & 15;                      // A-fragment row
    const int kb = (lane >> 4) * 8;                // A-fragment K base

    for (int s = 0; s < SEQ; ++s) {
        // ---- Phase A0: per-row 5-dim features ----
        if (tid < ROWS_PER_BLOCK) {
            int r = rowbase + tid;
            float x  = lanes[(size_t)(s * BLANES + r) * 2 + 0];
            float y  = lanes[(size_t)(s * BLANES + r) * 2 + 1];
            float px = (s > 0) ? lanes[(size_t)((s - 1) * BLANES + r) * 2 + 0] : x;
            float py = (s > 0) ? lanes[(size_t)((s - 1) * BLANES + r) * 2 + 1] : y;
            float yaw = atan2f(y, x);
            int ok = valid_s[tid];
            feat_s[tid][0] = ok ? x : 0.0f;
            feat_s[tid][1] = ok ? y : 0.0f;
            feat_s[tid][2] = ok ? (x - px) : 0.0f;
            feat_s[tid][3] = ok ? (y - py) : 0.0f;
            feat_s[tid][4] = ok ? yaw : 0.0f;
            if (s + 1 < SEQ)   // warm L2/L0 for next step's coordinates
                __builtin_prefetch(&lanes[(size_t)((s + 1) * BLANES + r) * 2], 0, 0);
        }
        __syncthreads();

        // ---- Phase A1: emb = relu(feat @ Wemb^T + bemb) -> bf16 LDS tile ----
#pragma unroll
        for (int i = 0; i < 8; ++i) {
            int e   = tid * 8 + i;                 // 2048 entries / 256 threads
            int row = e >> 7;
            int hc  = e & 127;
            const float* wv = &wemb_s[hc * 5];
            float v = bemb_s[hc]
                    + wv[0] * feat_s[row][0] + wv[1] * feat_s[row][1]
                    + wv[2] * feat_s[row][2] + wv[3] * feat_s[row][3]
                    + wv[4] * feat_s[row][4];
            emb_s[row][hc] = (__bf16)fmaxf(v, 0.0f);
        }
        __syncthreads();

        // ---- Phase B: gates = bias + emb@Wih^T + h@Whh^T via bf16 WMMA ----
        v8f acc[4];
#pragma unroll
        for (int q = 0; q < 4; ++q) {
            v8f a = {bias[q], bias[q], bias[q], bias[q],
                     bias[q], bias[q], bias[q], bias[q]};
            acc[q] = a;
        }
#pragma unroll
        for (int kc = 0; kc < KCHUNKS; ++kc) {
            v8bf xlo = *(const v8bf*)&emb_s[m][kc * 32 + kb];
            v8bf xhi = *(const v8bf*)&emb_s[m][kc * 32 + kb + 16];
            v8bf hlo = *(const v8bf*)&h_s[m][kc * 32 + kb];
            v8bf hhi = *(const v8bf*)&h_s[m][kc * 32 + kb + 16];
            v16bf ax, ah;
#pragma unroll
            for (int i = 0; i < 8; ++i) {
                ax[i] = xlo[i]; ax[i + 8] = xhi[i];
                ah[i] = hlo[i]; ah[i + 8] = hhi[i];
            }
#pragma unroll
            for (int q = 0; q < 4; ++q) {
                int tile = q * 8 + w;
                size_t fo = (size_t)((tile * KCHUNKS + kc) * 32 + lane) * 16;
                v16bf bx = *(const v16bf*)&frags[fo];
                v16bf bh = *(const v16bf*)&frags[FRAG_ELEMS_PER_MATRIX + fo];
                acc[q] = __builtin_amdgcn_wmma_f32_16x16x32_bf16(
                             false, ax, false, bx, (short)0, acc[q], false, false);
                acc[q] = __builtin_amdgcn_wmma_f32_16x16x32_bf16(
                             false, ah, false, bh, (short)0, acc[q], false, false);
            }
        }

        __syncthreads();   // everyone done reading h_s before we overwrite it

        // ---- Phase C: elementwise LSTM cell update ----
        int hc = w * 16 + (lane & 15);
#pragma unroll
        for (int j = 0; j < 8; ++j) {
            float iv = fast_sigmoid(acc[0][j]);
            float fv = fast_sigmoid(acc[1][j]);
            float gv = fast_tanh(acc[2][j]);
            float ov = fast_sigmoid(acc[3][j]);
            float c  = fv * cst[j] + iv * gv;
            cst[j]   = c;
            float hv = ov * fast_tanh(c);
            int rr   = j + ((lane >> 4) << 3);     // C-layout row
            h_s[rr][hc] = (__bf16)hv;
        }
        __syncthreads();
    }

    // ---- Epilogue: final h -> f32 output [BLANES][H] ----
    for (int i = tid; i < ROWS_PER_BLOCK * HDIM; i += 256) {
        int row = i >> 7;
        int hc  = i & 127;
        out[(size_t)(rowbase + row) * HDIM + hc] = (float)h_s[row][hc];
    }
}

extern "C" void kernel_launch(void* const* d_in, const int* in_sizes, int n_in,
                              void* d_out, int out_size, void* d_ws, size_t ws_size,
                              hipStream_t stream) {
    // setup_inputs() order:
    // 0: agent_past_motion_context (unused by reference math)
    // 1: possible_lanes [80, 2560, 2]
    // 2: W_emb [128,5]   3: b_emb [128]
    // 4: W_ih [512,128]  5: W_hh [512,128]
    // 6: b_ih [512]      7: b_hh [512]
    const float* lanes = (const float*)d_in[1];
    const float* Wemb  = (const float*)d_in[2];
    const float* bemb  = (const float*)d_in[3];
    const float* Wih   = (const float*)d_in[4];
    const float* Whh   = (const float*)d_in[5];
    const float* bih   = (const float*)d_in[6];
    const float* bhh   = (const float*)d_in[7];
    float* out = (float*)d_out;

    __bf16* frags = (__bf16*)d_ws;   // needs 2*65536*2 = 256 KB of scratch

    // Pack both weight matrices into WMMA-friendly bf16 fragments (L2-resident).
    lanefe_pack_weights<<<(2 * FRAG_ELEMS_PER_MATRIX) / 256, 256, 0, stream>>>(
        Wih, Whh, frags);

    // Persistent fused kernel: 2560 rows / 16 rows-per-block = 160 blocks.
    lanefe_lstm_kernel<<<BLANES / ROWS_PER_BLOCK, 256, 0, stream>>>(
        lanes, Wemb, bemb, bih, bhh, frags, out);
}
